// GeneralPolyGNN_52544629899590
// MI455X (gfx1250) — matrix operand order repaired and checked by reading, not compiled, not run on hardware
//
#include <hip/hip_runtime.h>

typedef __bf16 bf16;
typedef __attribute__((ext_vector_type(16))) __bf16 v16bf;
typedef __attribute__((ext_vector_type(8)))  __bf16 v8bf;
typedef __attribute__((ext_vector_type(8)))  float   v8f;
typedef __attribute__((ext_vector_type(4)))  unsigned int v4u;
typedef __attribute__((ext_vector_type(8)))  int          v8i;
typedef __attribute__((ext_vector_type(4)))  int          v4i;

#define NB 128            // graphs
#define NN 512            // nodes per graph
#define NF 128            // feature dim
#define NE (NB * NN * 16) // edges

// ---------------------------------------------------------------------------
// Tensor Data Mover: 2D tile (tile_d1 rows x tile_d0 elems of 2B) from a
// row-major matrix with row stride `stride0` elems -> contiguous LDS at lds_off.
// D# per CDNA5 ISA (group0: count/type/addr, group1: dims).
// ---------------------------------------------------------------------------
__device__ __forceinline__ unsigned lds_off_u32(const void* p) {
  // addrspace(3) -> flat keeps the LDS byte offset in the low 32 bits
  return (unsigned)(unsigned long long)(uintptr_t)p;
}

__device__ __forceinline__ void tdm_load_2d(unsigned lds_off, const void* gptr,
                                            unsigned tile_d0, unsigned tile_d1,
                                            unsigned long long stride0) {
  unsigned long long ga = (unsigned long long)(uintptr_t)gptr;
  v4u g0;
  g0[0] = 1u;                                               // count=1, is_restore=0
  g0[1] = lds_off;                                          // lds_addr [63:32]
  g0[2] = (unsigned)(ga & 0xffffffffu);                     // global_addr lo
  g0[3] = (unsigned)((ga >> 32) & 0x01ffffffu) | 0x80000000u; // addr[56:32] | type=2
  v8i g1;
  g1[0] = (int)0x00010000u;                                  // wg_mask=0, data_size=1 (2B)
  g1[1] = (int)((tile_d0 & 0xffffu) << 16);                  // tensor_dim0[15:0] @ [63:48]
  g1[2] = (int)(((tile_d0 >> 16) & 0xffffu) |                // tensor_dim0[31:16]
                ((tile_d1 & 0xffffu) << 16));                // tensor_dim1[15:0]
  g1[3] = (int)(((tile_d1 >> 16) & 0xffffu) |                // tensor_dim1[31:16]
                ((tile_d0 & 0xffffu) << 16));                // tile_dim0
  g1[4] = (int)(tile_d1 & 0xffffu);                          // tile_dim1 ; tile_dim2=0
  g1[5] = (int)(stride0 & 0xffffffffu);                      // dim0_stride lo
  g1[6] = (int)((stride0 >> 32) & 0xffffu);                  // dim0_stride hi ; dim1_stride lo=0
  g1[7] = 0;
  v4i z4 = {0, 0, 0, 0};
#if defined(__clang_major__) && (__clang_major__ >= 23)
  v8i z8 = {0, 0, 0, 0, 0, 0, 0, 0};
  __builtin_amdgcn_tensor_load_to_lds(g0, g1, z4, z4, z8, 0);
#else
  __builtin_amdgcn_tensor_load_to_lds(g0, g1, z4, z4, 0);
#endif
}

// ---------------------------------------------------------------------------
// Elementwise helpers
// ---------------------------------------------------------------------------
__global__ void k_zero16(float4* p, long long n16) {
  long long i = (long long)blockIdx.x * blockDim.x + threadIdx.x;
  if (i < n16) p[i] = make_float4(0.f, 0.f, 0.f, 0.f);
}

__global__ void k_scatter(const int* __restrict__ src, const int* __restrict__ dst,
                          bf16* __restrict__ S, int E) {
  int e = blockIdx.x * blockDim.x + threadIdx.x;
  if (e >= E) return;
  int s = src[e], d = dst[e];
  int g  = s >> 9;
  int si = s & 511;
  int di = d & 511;
  S[((long long)g * NN + si) * NN + di] = (bf16)1.0f;  // .set(1.0): dedup via store
}

// one wave per adjacency row: dinv = deg>0 ? rsqrt(deg) : 0
__global__ void k_degree(const bf16* __restrict__ S, float* __restrict__ dinv) {
  long long gwave = ((long long)blockIdx.x * blockDim.x + threadIdx.x) >> 5;
  int lane = threadIdx.x & 31;
  if (gwave >= (long long)NB * NN) return;
  const bf16* p = S + gwave * NN;
  float s = 0.f;
  for (int j = lane; j < NN; j += 32) s += (float)p[j];
  for (int off = 16; off > 0; off >>= 1) s += __shfl_down(s, off);
  if (lane == 0) dinv[gwave] = (s > 0.f) ? rsqrtf(s) : 0.f;
}

__global__ void k_scale(bf16* __restrict__ S, const float* __restrict__ dinv) {
  long long idx = (long long)blockIdx.x * blockDim.x + threadIdx.x;
  if (idx >= (long long)NB * NN * NN) return;
  int j = (int)(idx & (NN - 1));
  long long row = idx >> 9;   // g*NN + i
  long long g   = row >> 9;
  float v = (float)S[idx];
  S[idx] = (bf16)(v * dinv[row] * dinv[g * NN + j]);
}

__global__ void k_cvt(const float* __restrict__ x, bf16* __restrict__ y, long long n) {
  long long i = (long long)blockIdx.x * blockDim.x + threadIdx.x;
  if (i < n) y[i] = (bf16)x[i];
}

// transpose-convert W[f][o] -> Wt[o][f] in bf16 (128x128)
__global__ void k_cvt_t(const float* __restrict__ w, bf16* __restrict__ wt) {
  int i = blockIdx.x * blockDim.x + threadIdx.x;
  if (i >= NF * NF) return;
  int f = i >> 7, o = i & 127;
  wt[o * NF + f] = (bf16)w[f * NF + o];
}

// ---------------------------------------------------------------------------
// acc {=, +=} src ; zt[g][f][node] = bf16(src)  (LDS-tiled 32x32 transpose,
// coalesced reads and writes). src may alias acc when add==0.
// grid.x = NB * (NN/32) * (NF/32) = 8192, block = 256.
// ---------------------------------------------------------------------------
__global__ void k_stage(const float* src, float* acc, bf16* __restrict__ zt, int add) {
  __shared__ bf16 tile[32][33];
  int t  = blockIdx.x;
  int g  = t >> 6;
  int r  = t & 63;
  int n0 = (r >> 2) << 5;   // node tile base
  int f0 = (r & 3) << 5;    // feature tile base
  int col = threadIdx.x & 31;
  int row = threadIdx.x >> 5;   // 8 rows per pass
  const long long base = (long long)g * NN * NF;
#pragma unroll
  for (int p = 0; p < 4; ++p) {
    int n = n0 + row + p * 8;
    long long i = base + (long long)n * NF + f0 + col;
    float v = src[i];
    if (add) acc[i] += v; else acc[i] = v;
    tile[row + p * 8][col] = (bf16)v;
  }
  __syncthreads();
#pragma unroll
  for (int p = 0; p < 4; ++p) {
    int f = f0 + row + p * 8;
    zt[((long long)g * NF + f) * NN + n0 + col] = tile[col][row + p * 8];
  }
}

// ---------------------------------------------------------------------------
// WMMA bf16 GEMM: C[M x 128] = A[M x K] * B[K x 128], B given TRANSPOSED
// (Bt[128][K], row-major, row stride = K). TDM-staged, double-buffered LDS.
// grid.x = M/128, grid.y = batch; 256 threads = 8 waves.
// ---------------------------------------------------------------------------
__global__ __launch_bounds__(256)
void k_wmma_gemm(const bf16* __restrict__ A, const bf16* __restrict__ Bt,
                 float* __restrict__ C, int K,
                 long long strideA, long long strideB, long long strideC,
                 const float* __restrict__ bias, int relu)
{
  __shared__ bf16 sA[2][128 * 32];   // [row m 0..127][k 0..31]
  __shared__ bf16 sB[2][128 * 32];   // [row n 0..127][k 0..31]

  A  += (long long)blockIdx.y * strideA;
  Bt += (long long)blockIdx.y * strideB;
  C  += (long long)blockIdx.y * strideC;

  const int tid   = threadIdx.x;
  const int wave  = tid >> 5;
  const int lane  = tid & 31;
  const int lrow  = lane & 15;
  const int khalf = lane >> 4;
  const long long mBase = (long long)blockIdx.x * 128;
  const bf16* Abase = A + mBase * (long long)K;
  const int nk = K >> 5;

  if (wave == 0) {   // prologue: chunk 0 into buffer 0 (one wave drives the TDM)
    tdm_load_2d(lds_off_u32(sA[0]), Abase, 32, 128, (unsigned long long)K);
    tdm_load_2d(lds_off_u32(sB[0]), Bt,    32, 128, (unsigned long long)K);
  }

  v8f acc[8] = {};

  for (int i = 0; i < nk; ++i) {
    const int kc = i << 5;
    if (wave == 0) {
      if (i + 1 < nk) {   // prefetch next chunk into the other buffer
        tdm_load_2d(lds_off_u32(sA[(i + 1) & 1]), Abase + kc + 32, 32, 128,
                    (unsigned long long)K);
        tdm_load_2d(lds_off_u32(sB[(i + 1) & 1]), Bt + kc + 32, 32, 128,
                    (unsigned long long)K);
        __builtin_amdgcn_s_wait_tensorcnt(2);  // current chunk's 2 loads done
      } else {
        __builtin_amdgcn_s_wait_tensorcnt(0);
      }
    }
    __syncthreads();   // chunk i visible to all waves

    const bf16* cA = sA[i & 1];
    const bf16* cB = sB[i & 1];

    // A fragment (16x32 bf16): lane<16 -> K {0..7,16..23}; lane>=16 -> {8..15,24..31}
    union { v16bf v; v8bf h[2]; } fa;
    const bf16* pa = cA + (wave * 16 + lrow) * 32 + khalf * 8;
    fa.h[0] = *(const v8bf*)pa;
    fa.h[1] = *(const v8bf*)(pa + 16);

#pragma unroll
    for (int t = 0; t < 8; ++t) {
      // B fragment (32x16): lane<16 -> K {0..7,16..23}; lane>=16 -> {8..15,24..31}
      union { v16bf v; v8bf h[2]; } ub;
      const bf16* pb = cB + (t * 16 + lrow) * 32 + khalf * 8;
      ub.h[0] = *(const v8bf*)pb;
      ub.h[1] = *(const v8bf*)(pb + 16);
      v16bf fb = ub.v;
      acc[t] = __builtin_amdgcn_wmma_f32_16x16x32_bf16(
          false, fa.v, false, fb, (short)0, acc[t], false, false);
    }
    __syncthreads();   // all reads of this buffer done before it is re-staged
  }

  // epilogue: C 16x16 f32 layout: n = lrow, m = khalf*8 + r
#pragma unroll
  for (int t = 0; t < 8; ++t) {
    int n = t * 16 + lrow;
    float bv = bias ? bias[n] : 0.f;
#pragma unroll
    for (int r = 0; r < 8; ++r) {
      long long m = mBase + wave * 16 + khalf * 8 + r;
      float v = acc[t][r] + bv;
      if (relu) v = fmaxf(v, 0.f);
      C[m * 128 + n] = v;
    }
  }
}

// ---------------------------------------------------------------------------
// Pool + head
// ---------------------------------------------------------------------------
__global__ void k_pool(const float* __restrict__ x, float* __restrict__ h) {
  int i = blockIdx.x * blockDim.x + threadIdx.x;   // 16384 = B*F
  if (i >= NB * NF) return;
  int g = i >> 7, f = i & 127;
  const float* p = x + (long long)g * NN * NF + f;
  float s = 0.f;
  for (int n = 0; n < NN; ++n) s += p[(long long)n * NF];
  h[i] = s * (1.0f / NN);
}

__global__ void k_head(const float* __restrict__ h,
                       const float* __restrict__ Wr1, const float* __restrict__ br1,
                       const float* __restrict__ Wr2, const float* __restrict__ br2,
                       float* __restrict__ out) {
  int b = threadIdx.x;
  if (b >= NB) return;
  float o = 0.f;
  for (int j = 0; j < 64; ++j) {
    float t = br1[j];
    for (int f = 0; f < NF; ++f) t += h[b * NF + f] * Wr1[f * 64 + j];
    t = fmaxf(t, 0.f);
    o += t * Wr2[j];
  }
  out[b] = o + br2[0];
}

// ---------------------------------------------------------------------------
extern "C" void kernel_launch(void* const* d_in, const int* in_sizes, int n_in,
                              void* d_out, int out_size, void* d_ws, size_t ws_size,
                              hipStream_t stream) {
  const float* X   = (const float*)d_in[0];
  const int*   ei  = (const int*)d_in[2];   // [2, E] flat: src then dst
  const float* W1  = (const float*)d_in[3];
  const float* b1  = (const float*)d_in[4];
  const float* W2  = (const float*)d_in[5];
  const float* b2  = (const float*)d_in[6];
  const float* Wr1 = (const float*)d_in[7];
  const float* br1 = (const float*)d_in[8];
  const float* Wr2 = (const float*)d_in[9];
  const float* br2 = (const float*)d_in[10];
  float* out = (float*)d_out;

  char* ws = (char*)d_ws;
  const long long S_bytes = (long long)NB * NN * NN * 2;      // 64 MB bf16
  const long long nXF     = (long long)NB * NN * NF;          // 8,388,608
  bf16*  S     = (bf16*)(ws);
  bf16*  zt    = (bf16*)(ws + S_bytes);                       // 16 MB (transposed)
  float* znew  = (float*)(ws + S_bytes + nXF * 2);            // 32 MB
  float* acc   = (float*)(ws + S_bytes + nXF * 2 + nXF * 4);  // 32 MB (also layer out)
  bf16*  accbf = (bf16*)(ws + S_bytes + nXF * 2 + nXF * 8);   // 16 MB
  char*  tail  = ws + S_bytes + nXF * 2 + nXF * 8 + nXF * 2;
  float* dinv  = (float*)tail;                                // 256 KB
  bf16*  W1t   = (bf16*)(tail + 262144);
  bf16*  W2t   = (bf16*)(tail + 262144 + 32768);
  float* h     = (float*)(tail + 262144 + 65536);

  const long long sS  = (long long)NN * NN;   // per-graph S stride
  const long long sZt = (long long)NF * NN;   // per-graph transposed-Z stride
  const long long sZ  = (long long)NN * NF;   // per-graph C stride

  // ---- build normalized GSO S (bf16) ----
  k_zero16<<<16384, 256, 0, stream>>>((float4*)S, S_bytes / 16);
  k_scatter<<<NE / 256, 256, 0, stream>>>(ei, ei + NE, S, NE);
  k_degree<<<(NB * NN) / 8, 256, 0, stream>>>(S, dinv);
  k_scale<<<(int)(((long long)NB * NN * NN) / 256), 256, 0, stream>>>(S, dinv);
  k_cvt_t<<<64, 256, 0, stream>>>(W1, W1t);
  k_cvt_t<<<64, 256, 0, stream>>>(W2, W2t);

  // ---- layer 1: acc = sum_{k=0..3} S^k x ; x1 = relu(acc @ W1 + b1) ----
  k_stage<<<8192, 256, 0, stream>>>(X, acc, zt, 0);
  for (int it = 0; it < 3; ++it) {
    k_wmma_gemm<<<dim3(NN / 128, NB), 256, 0, stream>>>(S, zt, znew, NN,
                                                        sS, sZt, sZ, nullptr, 0);
    k_stage<<<8192, 256, 0, stream>>>(znew, acc, zt, 1);
  }
  k_cvt<<<32768, 256, 0, stream>>>(acc, accbf, nXF);
  k_wmma_gemm<<<dim3((NB * NN) / 128, 1), 256, 0, stream>>>(accbf, W1t, acc, NF,
                                                            0, 0, 0, b1, 1);

  // ---- layer 2 ----
  k_stage<<<8192, 256, 0, stream>>>(acc, acc, zt, 0);
  for (int it = 0; it < 3; ++it) {
    k_wmma_gemm<<<dim3(NN / 128, NB), 256, 0, stream>>>(S, zt, znew, NN,
                                                        sS, sZt, sZ, nullptr, 0);
    k_stage<<<8192, 256, 0, stream>>>(znew, acc, zt, 1);
  }
  k_cvt<<<32768, 256, 0, stream>>>(acc, accbf, nXF);
  k_wmma_gemm<<<dim3((NB * NN) / 128, 1), 256, 0, stream>>>(accbf, W2t, acc, NF,
                                                            0, 0, 0, b2, 1);

  // ---- pool + head ----
  k_pool<<<64, 256, 0, stream>>>(acc, h);
  k_head<<<1, 128, 0, stream>>>(h, Wr1, br1, Wr2, br2, out);
}
